// GDN_53781580480873
// MI455X (gfx1250) — compile-verified, matrix operand order checked by reading, not run on hardware
//
#include <hip/hip_runtime.h>

// GDN forward for MI455X (gfx1250, wave32, WMMA).
// Batched graph attention collapses to one dense 512x512 attention matrix
// (same topology + same logits for every batch) -> three bf16 WMMA GEMMs.
// Round 4: cheap bf16 packing via +0x8000 round + v_perm_b32 (cvt_pk_bf16_f32
// builtin is not declared on this toolchain; manual RNE chains dominated VALU).

#define Bn      128
#define Nn      512
#define F_INc   64
#define Dd      128
#define Ee      8192
#define BNROWS  65536
#define EPSc    1e-5f

typedef __attribute__((ext_vector_type(16))) __bf16        v16bf;
typedef __attribute__((ext_vector_type(8)))  float         v8f;
typedef __attribute__((ext_vector_type(8)))  unsigned int  v8u;
typedef int v4i_vs __attribute__((vector_size(16)));

#if defined(__has_builtin)
#if __has_builtin(__builtin_amdgcn_cvt_pk_bf16_f32)
#define USE_CVT_PK 1
#endif
#if __has_builtin(__builtin_amdgcn_perm)
#define USE_PERM 1
#endif
#if __has_builtin(__builtin_amdgcn_global_load_async_to_lds_b128) && \
    __has_builtin(__builtin_amdgcn_s_wait_asynccnt)
#define USE_ASYNC_LDS 1
#endif
#endif

// f32 -> bf16 pair, round-to-nearest (half-up): bits+0x8000 then take hi16.
// v_perm_b32 merges both high halves in one instruction.
__device__ __forceinline__ unsigned int pack2(float lo, float hi) {
#ifdef USE_CVT_PK
  return __builtin_bit_cast(unsigned int, __builtin_amdgcn_cvt_pk_bf16_f32(lo, hi));
#else
  unsigned int ulo = __builtin_bit_cast(unsigned int, lo) + 0x8000u;
  unsigned int uhi = __builtin_bit_cast(unsigned int, hi) + 0x8000u;
#ifdef USE_PERM
  return __builtin_amdgcn_perm(uhi, ulo, 0x07060302u);  // {uhi[31:16], ulo[31:16]}
#else
  return (ulo >> 16) | (uhi & 0xFFFF0000u);
#endif
#endif
}
__device__ __forceinline__ unsigned short f2bf(float f) {
#ifdef USE_CVT_PK
  return (unsigned short)pack2(f, 0.f);
#else
  return (unsigned short)((__builtin_bit_cast(unsigned int, f) + 0x8000u) >> 16);
#endif
}

// Copy 16 KB from global to LDS, whole 256-thread block cooperating (64 B/thread).
// Uses gfx1250 async-to-LDS DMA (ASYNCcnt) — confirmed live on this toolchain.
__device__ __forceinline__ void stage_to_lds_16k(const void* gsrc, void* lds) {
  const char* g = (const char*)gsrc;
  char* l = (char*)lds;
  int off = threadIdx.x * 16;
#ifdef USE_ASYNC_LDS
  typedef __attribute__((address_space(1))) v4i_vs glb_v4i;
  typedef __attribute__((address_space(3))) v4i_vs lds_v4i;
  #pragma unroll
  for (int j = 0; j < 4; j++) {
    __builtin_amdgcn_global_load_async_to_lds_b128(
        (glb_v4i*)(g + off + j * 4096),
        (lds_v4i*)(l + off + j * 4096), 0, 0);
  }
  __builtin_amdgcn_s_wait_asynccnt(0);
#else
  #pragma unroll
  for (int j = 0; j < 4; j++)
    *(uint4*)(l + off + j * 4096) = *(const uint4*)(g + off + j * 4096);
#endif
  __syncthreads();
}

// WMMA 16x16x32 bf16 operand fragment from a row-major bf16 matrix
// (uint pairs). Per CDNA5 ISA 16-bit A/B layout:
//   lanes 0-15 : row = lane,    K chunks {k0..k0+7} and {k0+16..k0+23}
//   lanes 16-31: row = lane-16, same chunks shifted by +8
// ld (halfwords) and k0 multiples of 8 -> two 16B-aligned b128 loads
// (global_load_b128 from global memory, ds_load_b128 from LDS).
__device__ __forceinline__ v16bf load_frag_bf16(const unsigned int* mat, int row,
                                                int ld, int k0, int lane) {
  int kb = k0 + ((lane >> 4) & 1) * 8;
  const uint4* p = reinterpret_cast<const uint4*>(mat + ((row * ld + kb) >> 1));
  uint4 a = p[0];
  uint4 b = p[2];                      // +16 halfwords
  v8u u;
  u[0]=a.x; u[1]=a.y; u[2]=a.z; u[3]=a.w;
  u[4]=b.x; u[5]=b.y; u[6]=b.z; u[7]=b.w;
  return __builtin_bit_cast(v16bf, u);
}

// Same fragment, converting from an f32 row-major source on the fly.
__device__ __forceinline__ v16bf load_frag_f32cvt(const float* mat, int row,
                                                  int ld, int k0, int lane) {
  int kb = k0 + ((lane >> 4) & 1) * 8;
  const float4* p = reinterpret_cast<const float4*>(mat + (size_t)row * ld + kb);
  float4 a = p[0], b = p[1];
  float4 c = p[4], d = p[5];           // +16 floats
  v8u u;
  u[0]=pack2(a.x,a.y); u[1]=pack2(a.z,a.w);
  u[2]=pack2(b.x,b.y); u[3]=pack2(b.z,b.w);
  u[4]=pack2(c.x,c.y); u[5]=pack2(c.z,c.w);
  u[6]=pack2(d.x,d.y); u[7]=pack2(d.z,d.w);
  return __builtin_bit_cast(v16bf, u);
}

// ---- 1) edge multiplicity counts (topology identical across batches) ----
__global__ void k_edges(const int* __restrict__ ei, unsigned int* __restrict__ C) {
  int e = blockIdx.x * blockDim.x + threadIdx.x;
  if (e < Ee) {
    int s = ei[e], d = ei[Ee + e];
    atomicAdd(&C[d * Nn + s], 1u);
  } else if (e < Ee + Nn) {
    int n = e - Ee;                    // self loop
    atomicAdd(&C[n * Nn + n], 1u);
  }
}

// ---- 2) per-dst-row leaky-relu + multiplicity-weighted softmax -> dense A (bf16) ----
__global__ void k_softmax(const float* __restrict__ wa, const unsigned int* __restrict__ C,
                          unsigned short* __restrict__ Abf) {
  __shared__ float red[256];
  int r = blockIdx.x, t = threadIdx.x;
  unsigned int cnt[2]; float l[2];
  float lmax = -3.4e38f;
  #pragma unroll
  for (int j = 0; j < 2; j++) {
    int c = t + j * 256;
    cnt[j] = C[r * Nn + c];
    float w = wa[r * Nn + c];
    l[j] = (w > 0.f) ? w : 0.2f * w;
    if (cnt[j] > 0u) lmax = fmaxf(lmax, l[j]);
  }
  red[t] = lmax; __syncthreads();
  for (int s = 128; s > 0; s >>= 1) { if (t < s) red[t] = fmaxf(red[t], red[t + s]); __syncthreads(); }
  float mx = red[0]; __syncthreads();
  float ez[2]; float se = 0.f;
  #pragma unroll
  for (int j = 0; j < 2; j++) {
    ez[j] = cnt[j] ? (float)cnt[j] * __expf(l[j] - mx) : 0.f;
    se += ez[j];
  }
  red[t] = se; __syncthreads();
  for (int s = 128; s > 0; s >>= 1) { if (t < s) red[t] += red[t + s]; __syncthreads(); }
  float inv = 1.f / red[0];
  #pragma unroll
  for (int j = 0; j < 2; j++) Abf[r * Nn + t + j * 256] = f2bf(ez[j] * inv);
}

// ---- 3) transpose small weights to bf16 (so B-fragment loads are contiguous) ----
__global__ void k_prepw(const float* __restrict__ lin_w, const float* __restrict__ w_rec,
                        unsigned short* __restrict__ WT, unsigned short* __restrict__ WrT) {
  int i = blockIdx.x * blockDim.x + threadIdx.x;     // 8192 threads
  if (i < Dd * F_INc) {
    int d = i >> 6, k = i & 63;
    WT[d * F_INc + k] = f2bf(lin_w[k * Dd + d]);     // WT[d][k] = lin_w[k][d]
    int n = i >> 7, kk = i & 127;
    WrT[n * Dd + kk] = f2bf(w_rec[kk * F_INc + n]);  // WrT[n][k] = w_recons[k][n]
  }
}

// ---- 4) hT[b][d][n] = (x W)^T, via WMMA: (W^T)[128x64] @ (x^T)[64x512] ----
__global__ void k_lin(const float* __restrict__ x, const unsigned short* __restrict__ WT,
                      unsigned short* __restrict__ hT) {
  int lane = threadIdx.x & 31, wid = threadIdx.x >> 5;
  int t = blockIdx.x * 8 + wid;        // 32768 tiles
  int b = t >> 8, r = t & 255;
  int mt = r >> 5, nt = r & 31;        // 8 d-tiles x 32 node-tiles
  int rl = lane & 15, half = (lane >> 4) & 1;
  const unsigned int* WTu = (const unsigned int*)WT;
  v8f c = {};
  #pragma unroll
  for (int k0 = 0; k0 < F_INc; k0 += 32) {
    v16bf a = load_frag_bf16(WTu, mt * 16 + rl, F_INc, k0, lane);
    v16bf bf = load_frag_f32cvt(x, b * Nn + nt * 16 + rl, F_INc, k0, lane);
    c = __builtin_amdgcn_wmma_f32_16x16x32_bf16(false, a, false, bf, (short)0, c, false, false);
  }
  unsigned short* dst = hT + ((size_t)(b * Dd + mt * 16 + half * 8)) * Nn + nt * 16 + rl;
  #pragma unroll
  for (int i = 0; i < 8; i++) dst[(size_t)i * Nn] = f2bf(c[i]);
}

// ---- 5) out[b] = A @ h[b] + bias, accumulate BN sums ----
// All 8 waves of a block share the same 16x512 A-tile (same mt): stage it
// once into LDS asynchronously, then read fragments with ds_load_b128.
__global__ void k_agg(const unsigned short* __restrict__ Abf, const unsigned short* __restrict__ hT,
                      const float* __restrict__ gbias, float* __restrict__ outf,
                      float* __restrict__ bnsum) {
  __shared__ __align__(16) unsigned short Atile[16 * Nn];   // 16 KB
  int lane = threadIdx.x & 31, wid = threadIdx.x >> 5;
  int t = blockIdx.x * 8 + wid;        // 32768 tiles
  int b = t >> 8, r = t & 255;
  int mt = r >> 3, nt = r & 7;         // 32 dst-tiles x 8 feature-tiles
  int rl = lane & 15, half = (lane >> 4) & 1;

  stage_to_lds_16k(Abf + (size_t)mt * 16 * Nn, Atile);

  const unsigned int* Alds = (const unsigned int*)Atile;
  const unsigned int* Hu = (const unsigned int*)hT + (size_t)b * (Dd * Nn / 2);
  v8f c = {};
  for (int k0 = 0; k0 < Nn; k0 += 32) {
    __builtin_prefetch(Hu + (((nt * 16 + rl) * Nn + k0 + 64) >> 1), 0, 1);
    v16bf a = load_frag_bf16(Alds, rl, Nn, k0, lane);               // from LDS
    v16bf bf = load_frag_bf16(Hu, nt * 16 + rl, Nn, k0, lane);      // from global
    c = __builtin_amdgcn_wmma_f32_16x16x32_bf16(false, a, false, bf, (short)0, c, false, false);
  }
  int n = nt * 16 + rl;
  float bias = gbias[n];
  float s = 0.f, s2 = 0.f;
  float* dst = outf + ((size_t)(b * Nn + mt * 16 + half * 8)) * Dd + n;
  #pragma unroll
  for (int i = 0; i < 8; i++) {
    float v = c[i] + bias;
    dst[(size_t)i * Dd] = v;
    s += v; s2 += v * v;
  }
  atomicAdd(&bnsum[n], s);
  atomicAdd(&bnsum[Dd + n], s2);
}

// ---- 6) BN batch stats -> per-channel scale/shift ----
__global__ void k_bnfin(const float* __restrict__ bnsum, const float* __restrict__ gamma,
                        const float* __restrict__ beta, float* __restrict__ sscale) {
  int d = threadIdx.x;
  float mu  = bnsum[d] * (1.0f / BNROWS);
  float var = bnsum[Dd + d] * (1.0f / BNROWS) - mu * mu;
  float sc  = gamma[d] * rsqrtf(var + EPSc);
  sscale[d]      = sc;
  sscale[Dd + d] = beta[d] - mu * sc;
}

// ---- 7) fused BN+ReLU, recons GEMM (WMMA, 4 n-tiles), pred dot ----
// WrT (64x128 bf16 = 16 KB) is shared by every wave: async-stage it to LDS.
__global__ void k_head(const float* __restrict__ outf, const float* __restrict__ sscale,
                       const unsigned short* __restrict__ WrT, const float* __restrict__ w_pred,
                       const float* __restrict__ b_rec, const float* __restrict__ b_pred,
                       float* __restrict__ out_rec, float* __restrict__ out_pred) {
  __shared__ __align__(16) unsigned short Wtile[F_INc * Dd];  // 16 KB
  __shared__ float sS[Dd], sH[Dd], sWp[Dd];
  if (threadIdx.x < Dd) {
    sS[threadIdx.x]  = sscale[threadIdx.x];
    sH[threadIdx.x]  = sscale[Dd + threadIdx.x];
    sWp[threadIdx.x] = w_pred[threadIdx.x];
  }
  stage_to_lds_16k(WrT, Wtile);        // includes the block barrier

  int lane = threadIdx.x & 31, wid = threadIdx.x >> 5;
  int rt = blockIdx.x * 8 + wid;       // 4096 row tiles
  int R = rt * 16;
  int rl = lane & 15, half = (lane >> 4) & 1;
  const unsigned int* Wlds = (const unsigned int*)Wtile;
  v8f c0 = {}, c1 = {}, c2 = {}, c3 = {};
  float pred = 0.f;
  const float* rowp = outf + (size_t)(R + rl) * Dd;
  #pragma unroll
  for (int k0 = 0; k0 < Dd; k0 += 32) {
    int kb = k0 + half * 8;
    float z[16];
    #pragma unroll
    for (int j = 0; j < 16; j++) {
      int k = kb + (j < 8 ? j : 8 + j);          // {kb..kb+7, kb+16..kb+23}
      float v = rowp[k];
      float zz = fmaxf(v * sS[k] + sH[k], 0.f);  // BN + ReLU
      z[j] = zz;
      pred += zz * sWp[k];
    }
    v8u u;
    #pragma unroll
    for (int j = 0; j < 8; j++) u[j] = pack2(z[2 * j], z[2 * j + 1]);
    v16bf a = __builtin_bit_cast(v16bf, u);
    v16bf b0 = load_frag_bf16(Wlds,      rl, Dd, k0, lane);
    v16bf b1 = load_frag_bf16(Wlds, 16 + rl, Dd, k0, lane);
    v16bf b2 = load_frag_bf16(Wlds, 32 + rl, Dd, k0, lane);
    v16bf b3 = load_frag_bf16(Wlds, 48 + rl, Dd, k0, lane);
    c0 = __builtin_amdgcn_wmma_f32_16x16x32_bf16(false, a, false, b0, (short)0, c0, false, false);
    c1 = __builtin_amdgcn_wmma_f32_16x16x32_bf16(false, a, false, b1, (short)0, c1, false, false);
    c2 = __builtin_amdgcn_wmma_f32_16x16x32_bf16(false, a, false, b2, (short)0, c2, false, false);
    c3 = __builtin_amdgcn_wmma_f32_16x16x32_bf16(false, a, false, b3, (short)0, c3, false, false);
  }
  float br0 = b_rec[rl], br1 = b_rec[16 + rl], br2 = b_rec[32 + rl], br3 = b_rec[48 + rl];
  float* d0 = out_rec + (size_t)(R + half * 8) * F_INc + rl;
  #pragma unroll
  for (int i = 0; i < 8; i++) {
    float* row = d0 + (size_t)i * F_INc;
    row[0]  = c0[i] + br0;
    row[16] = c1[i] + br1;
    row[32] = c2[i] + br2;
    row[48] = c3[i] + br3;
  }
  pred += __shfl_xor(pred, 16, 32);    // combine k-halves (lanes L and L+16 share a row)
  if (lane < 16) out_pred[R + rl] = pred + b_pred[0];
}

extern "C" void kernel_launch(void* const* d_in, const int* in_sizes, int n_in,
                              void* d_out, int out_size, void* d_ws, size_t ws_size,
                              hipStream_t stream) {
  (void)in_sizes; (void)n_in; (void)out_size; (void)ws_size;
  const float* data   = (const float*)d_in[0];
  const int*   ei     = (const int*)d_in[1];
  const float* warr   = (const float*)d_in[2];
  const float* lin_w  = (const float*)d_in[3];
  const float* gbias  = (const float*)d_in[4];
  const float* gamma  = (const float*)d_in[5];
  const float* beta   = (const float*)d_in[6];
  const float* w_rec  = (const float*)d_in[7];
  const float* b_rec  = (const float*)d_in[8];
  const float* w_pred = (const float*)d_in[9];
  const float* b_pred = (const float*)d_in[10];

  char* ws = (char*)d_ws;
  unsigned int*   C    = (unsigned int*)ws;    ws += (size_t)Nn * Nn * 4;   // edge counts
  unsigned short* Abf  = (unsigned short*)ws;  ws += (size_t)Nn * Nn * 2;   // dense attention (bf16)
  unsigned short* WT   = (unsigned short*)ws;  ws += (size_t)Dd * F_INc * 2;
  unsigned short* WrT  = (unsigned short*)ws;  ws += (size_t)F_INc * Dd * 2;
  float*          bnsum= (float*)ws;           ws += 2 * Dd * 4;
  float*          sscl = (float*)ws;           ws += 2 * Dd * 4;
  ws = (char*)(((uintptr_t)ws + 255) & ~(uintptr_t)255);
  unsigned short* hT   = (unsigned short*)ws;  ws += (size_t)Bn * Dd * Nn * 2;  // 16 MB
  float*          outf = (float*)ws;           ws += (size_t)BNROWS * Dd * 4;   // 32 MB

  float* out_rec  = (float*)d_out;
  float* out_pred = out_rec + (size_t)BNROWS * F_INc;
  float* out_wa   = out_pred + BNROWS;

  (void)hipMemsetAsync(C, 0, (size_t)Nn * Nn * 4, stream);
  (void)hipMemsetAsync(bnsum, 0, 2 * Dd * 4, stream);

  k_edges  <<<(Ee + Nn + 255) / 256, 256, 0, stream>>>(ei, C);
  k_softmax<<<Nn, 256, 0, stream>>>(warr, C, Abf);
  k_prepw  <<<(Dd * F_INc + 255) / 256, 256, 0, stream>>>(lin_w, w_rec, WT, WrT);
  k_lin    <<<4096, 256, 0, stream>>>(data, WT, hT);
  k_agg    <<<4096, 256, 0, stream>>>(Abf, hT, gbias, outf, bnsum);
  k_bnfin  <<<1, Dd, 0, stream>>>(bnsum, gamma, beta, sscl);
  k_head   <<<512, 256, 0, stream>>>(outf, sscl, WrT, w_pred, b_rec, b_pred, out_rec, out_pred);

  (void)hipMemcpyAsync(out_wa, warr, (size_t)Nn * Nn * 4, hipMemcpyDeviceToDevice, stream);
}
